// SpatialSAGE_41575283426038
// MI455X (gfx1250) — matrix operand order repaired and checked by reading, not compiled, not run on hardware
//
#include <hip/hip_runtime.h>
#include <math.h>

#define N_NODES 100000
#define N_EDGES 1600000
#define IN_DIM  64
#define HID     128

typedef __attribute__((ext_vector_type(16))) _Float16 v16h;
typedef __attribute__((ext_vector_type(8)))  _Float16 v8h;
typedef __attribute__((ext_vector_type(8)))  float    v8f;

// ---------------------------------------------------------------------------
// A-fragment (16x32 f16 WMMA layout) from an fp32 source (atomic accumulators).
// lane L<16 : row = base+L,      elems 0..7 = K[k0..k0+7],   8..15 = K[k0+16..k0+23]
// lane L>=16: row = base+(L&15), elems 0..7 = K[k0+8..k0+15], 8..15 = K[k0+24..k0+31]
// ---------------------------------------------------------------------------
__device__ __forceinline__ v16h load_a_f32(const float* __restrict__ A, int ld,
                                           int row, int k0, int lane) {
    const int khalf = (lane >> 4) << 3;
    const float* p = A + (size_t)row * ld + k0 + khalf;
    v16h a;
#pragma unroll
    for (int j = 0; j < 8; ++j) a[j] = (_Float16)p[j];
#pragma unroll
    for (int j = 0; j < 8; ++j) a[j + 8] = (_Float16)p[j + 16];
    return a;
}

// Same fragment from a pre-converted f16 source: two 16B vector loads, no cvt.
__device__ __forceinline__ v16h load_a_f16(const _Float16* __restrict__ A, int ld,
                                           int row, int k0, int lane) {
    const int khalf = (lane >> 4) << 3;
    const _Float16* p = A + (size_t)row * ld + k0 + khalf;
    v8h lo = *(const v8h*)p;
    v8h hi = *(const v8h*)(p + 16);
    return __builtin_shufflevector(lo, hi, 0, 1, 2, 3, 4, 5, 6, 7,
                                   8, 9, 10, 11, 12, 13, 14, 15);
}

// B-fragment (32x16, B = W^T, W row-major [HID,K] in f16): same k order as A.
// lane L: col = colBase+(L&15), elems 0..7 = K[k0+(L>>4)*8+j], 8..15 = K[k0+16+(L>>4)*8+j]
__device__ __forceinline__ v16h load_b_f16(const _Float16* __restrict__ W, int ld,
                                           int col, int k0, int lane) {
    const int kb = k0 + ((lane >> 4) << 3);
    const _Float16* p = W + (size_t)col * ld + kb;
    v8h lo = *(const v8h*)p;
    v8h hi = *(const v8h*)(p + 16);
    return __builtin_shufflevector(lo, hi, 0, 1, 2, 3, 4, 5, 6, 7,
                                   8, 9, 10, 11, 12, 13, 14, 15);
}

__device__ __forceinline__ float elu(float v) {
    return v > 0.0f ? v : (__expf(v) - 1.0f);
}

// ---------------------------------------------------------------------------
// Fused SAGE layer: act( (agg @ Wl^T) * rdeg + bias + X @ Wr^T )
// Block = 256 threads = 8 waves; 16 rows x 128 cols per block, one col-tile
// per wave. rdeg (=1/max(deg,1)) is precomputed; epilogue is a single FMA.
// !HEAD: store f16 h-tile. HEAD: fuse ELU + (h @ Wh^T) row-reduction into outF.
// N_NODES % 16 == 0 -> EXEC all-ones for every WMMA.
// ---------------------------------------------------------------------------
template<int K, bool HEAD>
__global__ __launch_bounds__(256)
void sage_gemm(const float* __restrict__ Aagg, const _Float16* __restrict__ Aself,
               const _Float16* __restrict__ Wl, const _Float16* __restrict__ Wr,
               const float* __restrict__ bias,  const float* __restrict__ rdeg,
               const float* __restrict__ Wh,
               _Float16* __restrict__ outH, float* __restrict__ outF) {
    const int lane    = threadIdx.x & 31;
    const int wave    = threadIdx.x >> 5;
    const int rowBase = blockIdx.x << 4;
    const int colBase = wave << 4;
    const int arow    = rowBase + (lane & 15);
    const int col     = colBase + (lane & 15);

    v8f accA = {};   // agg path (scaled by rdeg in epilogue)
    v8f accS = {};   // self path
#pragma unroll
    for (int k0 = 0; k0 < K; k0 += 32) {
        v16h a = load_a_f32(Aagg, K, arow, k0, lane);
        v16h b = load_b_f16(Wl, K, col, k0, lane);
        accA = __builtin_amdgcn_wmma_f32_16x16x32_f16(false, a, false, b,
                                                      (short)0, accA, false, false);
    }
#pragma unroll
    for (int k0 = 0; k0 < K; k0 += 32) {
        v16h a = load_a_f16(Aself, K, arow, k0, lane);
        v16h b = load_b_f16(Wr, K, col, k0, lane);
        accS = __builtin_amdgcn_wmma_f32_16x16x32_f16(false, a, false, b,
                                                      (short)0, accS, false, false);
    }

    const int   rhalf = (lane >> 4) << 3;          // C/D row = rowBase + rhalf + r
    const float bcol  = bias[col];
    const float* rp   = rdeg + rowBase + rhalf;    // 8 consecutive rows per half

    if (!HEAD) {
#pragma unroll
        for (int r = 0; r < 8; ++r) {
            float v = elu(accA[r] * rp[r] + accS[r] + bcol);
            outH[(size_t)(rowBase + rhalf + r) * HID + col] = (_Float16)v;
        }
    } else {
        const float wh = Wh[col];
        float s[8];
#pragma unroll
        for (int r = 0; r < 8; ++r)
            s[r] = elu(accA[r] * rp[r] + accS[r] + bcol) * wh;
        // reduce over the 16 lanes of each half-wave (16 cols of this tile)
#pragma unroll
        for (int m = 1; m < 16; m <<= 1) {
#pragma unroll
            for (int r = 0; r < 8; ++r) s[r] += __shfl_xor(s[r], m, 32);
        }
        if ((lane & 15) == 0) {
#pragma unroll
            for (int r = 0; r < 8; ++r)
                atomicAdd(&outF[rowBase + rhalf + r], s[r]);
        }
    }
}

// ---------------------------------------------------------------------------
// Helpers
// ---------------------------------------------------------------------------
__global__ void zero_init(float* __restrict__ ws, float* __restrict__ out,
                          const float* __restrict__ bh, size_t nzero, int nout) {
    size_t i      = blockIdx.x * (size_t)blockDim.x + threadIdx.x;
    size_t stride = gridDim.x * (size_t)blockDim.x;
    const float b = bh[0];
    for (size_t j = i; j < nzero; j += stride) ws[j] = 0.0f;
    for (size_t j = i; j < (size_t)nout; j += stride) out[j] = b;
}

__global__ void cvt_f32_f16(const float* __restrict__ in, _Float16* __restrict__ out,
                            size_t n) {
    size_t i      = blockIdx.x * (size_t)blockDim.x + threadIdx.x;
    size_t stride = gridDim.x * (size_t)blockDim.x;
    for (size_t j = i; j < n; j += stride) out[j] = (_Float16)in[j];
}

__global__ void scatter_deg(const int* __restrict__ dst, float* __restrict__ deg, int E) {
    int e = blockIdx.x * blockDim.x + threadIdx.x;
    if (e < E) atomicAdd(&deg[dst[e]], 1.0f);
}

// deg -> rdeg in place: one precise division per node (O(N), trivial traffic).
__global__ void finalize_rdeg(float* __restrict__ deg, int n) {
    int i = blockIdx.x * blockDim.x + threadIdx.x;
    if (i < n) deg[i] = 1.0f / fmaxf(deg[i], 1.0f);
}

// Edge gather from f16 features (halves the dominant read stream),
// fp32 atomic accumulation for accuracy. 8 features per thread, 16B loads.
template<int D>
__global__ void scatter_feat16(const int* __restrict__ src, const int* __restrict__ dst,
                               const _Float16* __restrict__ X, float* __restrict__ agg,
                               int E) {
    constexpr int PER = D / 8;
    long long t = blockIdx.x * (long long)blockDim.x + threadIdx.x;
    int e = (int)(t / PER);
    int p = (int)(t % PER);
    if (e >= E) return;
    const v8h v = *(const v8h*)(X + (size_t)src[e] * D + (p << 3));
    float* o = agg + (size_t)dst[e] * D + (p << 3);
#pragma unroll
    for (int j = 0; j < 8; ++j) atomicAdd(o + j, (float)v[j]);
}

// ---------------------------------------------------------------------------
extern "C" void kernel_launch(void* const* d_in, const int* in_sizes, int n_in,
                              void* d_out, int out_size, void* d_ws, size_t ws_size,
                              hipStream_t stream) {
    (void)in_sizes; (void)n_in; (void)out_size; (void)ws_size;

    const float* x   = (const float*)d_in[0];
    const int*   ei  = (const int*)  d_in[1];
    const float* W1l = (const float*)d_in[2];
    const float* b1l = (const float*)d_in[3];
    const float* W1r = (const float*)d_in[4];
    const float* W2l = (const float*)d_in[5];
    const float* b2l = (const float*)d_in[6];
    const float* W2r = (const float*)d_in[7];
    const float* Wh  = (const float*)d_in[8];
    const float* bh  = (const float*)d_in[9];
    float*       out = (float*)d_out;

    const int* src = ei;
    const int* dst = ei + N_EDGES;

    const size_t N = N_NODES;
    float* ws   = (float*)d_ws;
    float* deg  = ws;                       // [N]   fp32 (becomes rdeg in place)
    float* agg1 = ws + N;                   // [N, 64]   fp32 (atomics)
    float* agg2 = ws + N + N * 64;          // [N, 128]  fp32 (atomics)
    const size_t nzero = N * 193;           // contiguous zero region

    _Float16* h1h  = (_Float16*)(ws + nzero);   // [N, 128] f16
    _Float16* xh   = h1h + N * (size_t)HID;     // [N, 64]  f16
    _Float16* w1lh = xh + N * (size_t)IN_DIM;   // [128, 64]
    _Float16* w1rh = w1lh + HID * IN_DIM;
    _Float16* w2lh = w1rh + HID * IN_DIM;       // [128, 128]
    _Float16* w2rh = w2lh + HID * HID;

    // 1) zero accumulators, seed out with bh
    zero_init<<<4096, 256, 0, stream>>>(ws, out, bh, nzero, N_NODES);

    // 2) one-time f16 conversions (x + the four weight matrices)
    cvt_f32_f16<<<4096, 256, 0, stream>>>(x,   xh,   N * (size_t)IN_DIM);
    cvt_f32_f16<<<32,   256, 0, stream>>>(W1l, w1lh, (size_t)HID * IN_DIM);
    cvt_f32_f16<<<32,   256, 0, stream>>>(W1r, w1rh, (size_t)HID * IN_DIM);
    cvt_f32_f16<<<64,   256, 0, stream>>>(W2l, w2lh, (size_t)HID * HID);
    cvt_f32_f16<<<64,   256, 0, stream>>>(W2r, w2rh, (size_t)HID * HID);

    // 3) in-degree, then deg -> rdeg = 1/max(deg,1) in place
    scatter_deg<<<(N_EDGES + 255) / 256, 256, 0, stream>>>(dst, deg, N_EDGES);
    finalize_rdeg<<<(N_NODES + 255) / 256, 256, 0, stream>>>(deg, N_NODES);

    // 4) layer-1 message scatter: agg1 += xh[src]
    {
        long long threads = (long long)N_EDGES * (IN_DIM / 8);
        scatter_feat16<IN_DIM><<<(int)((threads + 255) / 256), 256, 0, stream>>>(
            src, dst, xh, agg1, N_EDGES);
    }

    // 5) h1 (f16) = elu(agg1@W1l^T * rdeg + b1l + x@W1r^T)
    sage_gemm<IN_DIM, false><<<N_NODES / 16, 256, 0, stream>>>(
        agg1, xh, w1lh, w1rh, b1l, deg, nullptr, h1h, nullptr);

    // 6) layer-2 message scatter: agg2 += h1h[src]   (dominant stream, f16 reads)
    {
        long long threads = (long long)N_EDGES * (HID / 8);
        scatter_feat16<HID><<<(int)((threads + 255) / 256), 256, 0, stream>>>(
            src, dst, h1h, agg2, N_EDGES);
    }

    // 7) out += elu(agg2@W2l^T * rdeg + b2l + h1@W2r^T) @ Wh^T  (bh pre-seeded)
    sage_gemm<HID, true><<<N_NODES / 16, 256, 0, stream>>>(
        agg2, h1h, w2lh, w2rh, b2l, deg, Wh, nullptr, out);
}
